// MyModel_87522843558729
// MI455X (gfx1250) — matrix operand order repaired and checked by reading, not compiled, not run on hardware
//
#include <hip/hip_runtime.h>

// ---------------------------------------------------------------------------
// 2-layer LSTM + dense + softmax for MI455X (gfx1250), bf16 WMMA path.
// Batch rows are independent through the recurrence -> 8 persistent
// workgroups (16 batch rows each), full T loop inside the kernel.
// ---------------------------------------------------------------------------

typedef __attribute__((ext_vector_type(16))) __bf16 v16bf;
typedef __attribute__((ext_vector_type(8)))  __bf16 v8bf;
typedef __attribute__((ext_vector_type(8)))  float  v8f;

#define B_    128
#define T_    512
#define DIN   128
#define UNITS 512
#define NOUT  91
#define NGATE 2048   // 4*UNITS
#define NT_   128    // NGATE/16 n-tiles

// ---------------------------------------------------------------------------
// Pack [W;U] (f32, [K,2048] row-major each) into WMMA-B fragment-tiled bf16:
//   packed[((kt*128 + nt)*32 + lane)*16 + i]
// where per ISA B-layout (32x16 bf16): n = nt*16 + (lane&15),
//   k = kt*32 + (lane<16 ? 0 : 8) + (i<8 ? i : 8 + i).
// ---------------------------------------------------------------------------
__global__ void pack_wu_kernel(const float* __restrict__ W,
                               const float* __restrict__ U,
                               __bf16* __restrict__ out,
                               int in_k, int total) {
  int p = blockIdx.x * 256 + threadIdx.x;
  if (p >= total) return;
  int i    = p & 15;
  int lane = (p >> 4) & 31;
  int nt   = (p >> 9) & 127;
  int kt   = p >> 16;
  int n = nt * 16 + (lane & 15);
  int k = kt * 32 + ((lane < 16) ? 0 : 8) + ((i < 8) ? i : 8 + i);
  float v = (k < in_k) ? W[(long)k * NGATE + n]
                       : U[(long)(k - in_k) * NGATE + n];
  out[p] = (__bf16)v;
}

// ---------------------------------------------------------------------------
// Persistent LSTM layer. grid=8 (one 16-row batch block per WGP),
// block=1024 (32 waves). Wave w owns n-tiles {w, w+32, w+64, w+96} =
// gates i/f/g/o of units u = w*16 + (lane&15)  -> gating fully in registers.
// A = [x_t | h_{t-1}] lives in LDS as bf16; one GEMM [16,Kt]@[Kt,2048].
// ---------------------------------------------------------------------------
__launch_bounds__(1024, 1)
__global__ void lstm_layer_kernel(const void* __restrict__ in_seq,   // [B,T,in_k] f32 or bf16
                                  const __bf16* __restrict__ WU,     // packed [Kt/32][128][32][16]
                                  const float* __restrict__ bias,    // [2048]
                                  __bf16* __restrict__ h_out,        // [B,T,UNITS] bf16
                                  int in_k, int in_is_f32) {
  __shared__ __bf16 A_sh[16][1032];     // [16][in_k + 512 (+pad)] bf16

  const int tid  = threadIdx.x;
  const int lane = tid & 31;
  const int w    = tid >> 5;            // wave id 0..31
  const int b0   = blockIdx.x * 16;     // batch row base
  const int kt_total = (in_k + UNITS) / 32;

  // per-lane constants
  const int u      = w * 16 + (lane & 15);       // unit owned by this lane
  const float bi = bias[u];
  const float bf_g = bias[UNITS + u];
  const float bg = bias[2 * UNITS + u];
  const float bo = bias[3 * UNITS + u];
  const int koff   = (lane < 16) ? 0 : 8;        // A-fragment K sub-offset
  const int mrow   = lane & 15;                  // A-fragment row
  const int m_base = (lane < 16) ? 0 : 8;        // C-matrix row striping

  // staging / restage roles: 64 threads per row, 8 bf16 per thread
  const int sm = tid >> 6;
  const int sc = tid & 63;

  // zero initial h region
  {
    v8bf z = {};
    *(v8bf*)&A_sh[sm][in_k + sc * 8] = z;
  }
  float c[8];
#pragma unroll
  for (int r = 0; r < 8; ++r) c[r] = 0.f;

  for (int t = 0; t < T_; ++t) {
    // ---- stage input tile x_t (or h1_t) into A_sh[:, 0:in_k] ----
    if (in_is_f32) {                       // layer 1: in_k=128, 2 f32/thread
      const float* xp = (const float*)in_seq;
      float2 v = *(const float2*)&xp[((long)(b0 + sm) * T_ + t) * in_k + sc * 2];
      A_sh[sm][sc * 2 + 0] = (__bf16)v.x;
      A_sh[sm][sc * 2 + 1] = (__bf16)v.y;
    } else {                               // layer 2: in_k=512, 8 bf16/thread
      const __bf16* xp = (const __bf16*)in_seq;
      v8bf v = *(const v8bf*)&xp[((long)(b0 + sm) * T_ + t) * in_k + sc * 8];
      *(v8bf*)&A_sh[sm][sc * 8] = v;
    }
    __syncthreads();   // stage visible; prior-step h writes visible

    // ---- restage h_{t-1} to global (coalesced, one step behind) ----
    if (t > 0) {
      v8bf hv = *(const v8bf*)&A_sh[sm][in_k + sc * 8];
      *(v8bf*)&h_out[((long)(b0 + sm) * T_ + (t - 1)) * UNITS + sc * 8] = hv;
    }

    // ---- z = [x_t | h] @ [W;U] for this wave's 4 gate n-tiles ----
    v8f acc0 = {}, acc1 = {}, acc2 = {}, acc3 = {};
    for (int kt = 0; kt < kt_total; ++kt) {
      // A fragment per ISA 16-bit A layout: two contiguous 8-element runs
      v8bf lo = *(const v8bf*)&A_sh[mrow][kt * 32 + koff];
      v8bf hi = *(const v8bf*)&A_sh[mrow][kt * 32 + koff + 16];
      v16bf a;
#pragma unroll
      for (int i = 0; i < 8; ++i) { a[i] = lo[i]; a[8 + i] = hi[i]; }

      const __bf16* bp = WU + (long)kt * (NT_ * 512) + (long)lane * 16;
      v16bf bf0 = *(const v16bf*)(bp + (long)(w      ) * 512);
      acc0 = __builtin_amdgcn_wmma_f32_16x16x32_bf16(false, a, false, bf0, (short)0, acc0, false, false);
      v16bf bf1 = *(const v16bf*)(bp + (long)(w + 32 ) * 512);
      acc1 = __builtin_amdgcn_wmma_f32_16x16x32_bf16(false, a, false, bf1, (short)0, acc1, false, false);
      v16bf bf2 = *(const v16bf*)(bp + (long)(w + 64 ) * 512);
      acc2 = __builtin_amdgcn_wmma_f32_16x16x32_bf16(false, a, false, bf2, (short)0, acc2, false, false);
      v16bf bf3 = *(const v16bf*)(bp + (long)(w + 96 ) * 512);
      acc3 = __builtin_amdgcn_wmma_f32_16x16x32_bf16(false, a, false, bf3, (short)0, acc3, false, false);
    }
    __syncthreads();   // all A_sh reads consumed before h overwrite

    // ---- gates + cell update, write new h to LDS ----
#pragma unroll
    for (int r = 0; r < 8; ++r) {
      float zi = acc0[r] + bi;
      float zf = acc1[r] + bf_g;
      float zg = acc2[r] + bg;
      float zo = acc3[r] + bo;
      float ig = 0.5f * tanhf(0.5f * zi) + 0.5f;   // sigmoid
      float fg = 0.5f * tanhf(0.5f * zf) + 0.5f;
      float gg = tanhf(zg);
      float og = 0.5f * tanhf(0.5f * zo) + 0.5f;
      c[r] = fg * c[r] + ig * gg;
      float h = og * tanhf(c[r]);
      A_sh[m_base + r][in_k + u] = (__bf16)h;
    }
  }
  __syncthreads();
  // final h_{T-1}
  {
    v8bf hv = *(const v8bf*)&A_sh[sm][in_k + sc * 8];
    *(v8bf*)&h_out[((long)(b0 + sm) * T_ + (T_ - 1)) * UNITS + sc * 8] = hv;
  }
}

// ---------------------------------------------------------------------------
// Dense [512->91] + softmax. One block (128 threads) per (b,t) row.
// ---------------------------------------------------------------------------
__global__ void dense_softmax_kernel(const __bf16* __restrict__ h2,
                                     const float* __restrict__ Wd,  // [512,91]
                                     const float* __restrict__ bd,  // [91]
                                     float* __restrict__ pred,
                                     float* __restrict__ feat) {
  __shared__ float hrow[UNITS];
  __shared__ float red[128];
  const long row = blockIdx.x;
  const int tid = threadIdx.x;

#pragma unroll
  for (int i = 0; i < 4; ++i)
    hrow[tid + 128 * i] = (float)h2[row * UNITS + tid + 128 * i];
  __syncthreads();

  float zval = 0.f;
  float zmax = -1e30f;
  if (tid < NOUT) {
    float acc = bd[tid];
    for (int k = 0; k < UNITS; ++k)
      acc += hrow[k] * Wd[k * NOUT + tid];   // consecutive tids -> coalesced
    zval = acc;
    zmax = acc;
  }
  red[tid] = zmax;
  __syncthreads();
  for (int s = 64; s > 0; s >>= 1) {
    if (tid < s) red[tid] = fmaxf(red[tid], red[tid + s]);
    __syncthreads();
  }
  float mx = red[0];
  __syncthreads();
  float e = (tid < NOUT) ? __expf(zval - mx) : 0.f;
  red[tid] = e;
  __syncthreads();
  for (int s = 64; s > 0; s >>= 1) {
    if (tid < s) red[tid] += red[tid + s];
    __syncthreads();
  }
  float sum = red[0];
  if (tid < NOUT) {
    feat[row * NOUT + tid] = zval;
    pred[row * NOUT + tid] = e / sum;
  }
}

// ---------------------------------------------------------------------------
extern "C" void kernel_launch(void* const* d_in, const int* in_sizes, int n_in,
                              void* d_out, int out_size, void* d_ws, size_t ws_size,
                              hipStream_t stream) {
  const float* x  = (const float*)d_in[0];
  const float* W1 = (const float*)d_in[1];
  const float* U1 = (const float*)d_in[2];
  const float* b1 = (const float*)d_in[3];
  const float* W2 = (const float*)d_in[4];
  const float* U2 = (const float*)d_in[5];
  const float* b2 = (const float*)d_in[6];
  const float* Wd = (const float*)d_in[7];
  const float* bd = (const float*)d_in[8];
  float* out = (float*)d_out;

  char* ws = (char*)d_ws;
  const size_t off_wu1 = 0;                                  // 640*2048*2  = 2.50 MB
  const size_t off_wu2 = off_wu1 + (size_t)640 * 2048 * 2;   // 1024*2048*2 = 4.00 MB
  const size_t off_h1  = off_wu2 + (size_t)1024 * 2048 * 2;  // 128*512*512*2 = 64 MB
  const size_t off_h2  = off_h1 + (size_t)B_ * T_ * UNITS * 2;
  __bf16* wu1 = (__bf16*)(ws + off_wu1);
  __bf16* wu2 = (__bf16*)(ws + off_wu2);
  __bf16* h1  = (__bf16*)(ws + off_h1);
  __bf16* h2  = (__bf16*)(ws + off_h2);

  // pack [W;U] into WMMA-B fragment layout (bf16)
  {
    int total1 = 640 * 2048;
    pack_wu_kernel<<<(total1 + 255) / 256, 256, 0, stream>>>(W1, U1, wu1, DIN, total1);
    int total2 = 1024 * 2048;
    pack_wu_kernel<<<(total2 + 255) / 256, 256, 0, stream>>>(W2, U2, wu2, UNITS, total2);
  }

  // layer 1: x (f32) -> h1 (bf16)
  lstm_layer_kernel<<<8, 1024, 0, stream>>>(x, wu1, b1, h1, DIN, 1);
  // layer 2: h1 (bf16) -> h2 (bf16)
  lstm_layer_kernel<<<8, 1024, 0, stream>>>(h1, wu2, b2, h2, UNITS, 0);

  // dense + softmax: pred first, then pred_feat (tuple return order)
  float* pred = out;
  float* feat = out + (long)B_ * T_ * NOUT;
  dense_softmax_kernel<<<B_ * T_, 128, 0, stream>>>(h2, Wd, bd, pred, feat);
}
